// DeformableTransformerEncoderLayer_72172630442093
// MI455X (gfx1250) — matrix operand order repaired and checked by reading, not compiled, not run on hardware
//
#include <hip/hip_runtime.h>
#include <hip/hip_bf16.h>

typedef __attribute__((ext_vector_type(16))) _Float16 v16h;
typedef __attribute__((ext_vector_type(8)))  float    v8f;

#define CAMS 6
#define NQ   6400
#define LTOT 7979
#define DM   256
#define DFFN 1024

// ---------------------------------------------------------------------------
// Weight swizzle: f32 row-major (K x N) -> f16 WMMA B-fragments.
// Fragment (kt,nt) holds a 32x16 tile; per lane L: col = L&15, k = (L>>4)*16 + e
// stored contiguously: frag[(((kt*ntn + nt)*32 + L)*16 + e)]
// ---------------------------------------------------------------------------
__global__ void swizzle_w_kernel(const float* __restrict__ W, _Float16* __restrict__ frag,
                                 int K, int N) {
    size_t idx = (size_t)blockIdx.x * blockDim.x + threadIdx.x;
    size_t total = (size_t)K * N;
    if (idx >= total) return;
    int e    = (int)(idx & 15);
    int lane = (int)((idx >> 4) & 31);
    size_t f = idx >> 9;
    int ntn  = N >> 4;
    int nt   = (int)(f % ntn);
    int kt   = (int)(f / ntn);
    int k = kt * 32 + (lane >> 4) * 16 + e;
    int n = nt * 16 + (lane & 15);
    frag[idx] = (_Float16)W[(size_t)k * N + n];
}

// ---------------------------------------------------------------------------
// WMMA GEMM, 16 x (NACC*16) strip per wave: C[M x N] = A[M x K](f32)*Bfrag+bias.
// One A fragment per k-step feeds NACC independent WMMAs; with NACC*16 == N,
// every A element is loaded & converted exactly once (A is the dominant HBM
// stream: the 49MB value tensor). __launch_bounds__(256,1) unlocks the full
// CDNA5 VGPR file so the NACC=16 accumulator set (128 VGPRs) lives entirely
// in registers (no scratch spills); the wide strip's 16 independent WMMAs
// provide the ILP that occupancy would otherwise supply.
// A fragment (16x32 f16): lane L row = L&15; k = (L>>4)*16 + e
// C/D (16x16 f32): lane col = L&15; vgpr r -> row = r + 8*(L>>4)
// ---------------------------------------------------------------------------
template <int NACC>
__global__ void __launch_bounds__(256, 1)
wmma_gemm_kernel(const float* __restrict__ A,
                 const _Float16* __restrict__ Bfrag,
                 const float* __restrict__ bias,
                 float* __restrict__ C,
                 int M, int N, int K, int relu) {
    int lane = threadIdx.x & 31;
    int wave = threadIdx.x >> 5;
    int wpb  = blockDim.x >> 5;
    long tile = (long)blockIdx.x * wpb + wave;     // strip index
    int  ntn    = N >> 4;                          // 16-col fragments per k-row
    int  nstrip = N / (NACC * 16);                 // strips per row-band
    long mtiles = (M + 15) >> 4;
    if (tile >= mtiles * nstrip) return;           // wave-uniform exit
    int mt = (int)(tile / nstrip);
    int ns = (int)(tile % nstrip);
    int m_base = mt * 16;

    int row = m_base + (lane & 15);
    if (row >= M) row = M - 1;                     // clamp loads; stores guarded
    int half = lane >> 4;
    const float* arow = A + (size_t)row * K;

    int nkt = K >> 5;
    v8f acc[NACC];
#pragma unroll
    for (int j = 0; j < NACC; ++j) acc[j] = (v8f){};

    for (int kt = 0; kt < nkt; ++kt) {
        int k0 = kt * 32 + half * 16;
        if (kt + 1 < nkt)
            __builtin_prefetch(arow + k0 + 32, 0, 1);   // global_prefetch_b8
        v16h a;
#pragma unroll
        for (int e = 0; e < 8; ++e) a[e]     = (_Float16)arow[k0 + e];
#pragma unroll
        for (int e = 0; e < 8; ++e) a[8 + e] = (_Float16)arow[k0 + 8 + e];

        const _Float16* bk = Bfrag + (((size_t)kt * ntn + ns * NACC) * 32 + lane) * 16;
#pragma unroll
        for (int j = 0; j < NACC; ++j) {
            v16h b = *(const v16h*)(bk + (size_t)j * 512);
            acc[j] = __builtin_amdgcn_wmma_f32_16x16x32_f16(
                false, a, false, b, (short)0, acc[j], false, false);
        }
    }

#pragma unroll
    for (int j = 0; j < NACC; ++j) {
        int col = (ns * NACC + j) * 16 + (lane & 15);
        float bv = bias ? bias[col] : 0.0f;
#pragma unroll
        for (int r = 0; r < 8; ++r) {
            int mr = m_base + r + half * 8;
            if (mr < M) {
                float v = acc[j][r] + bv;
                if (relu) v = fmaxf(v, 0.0f);
                C[(size_t)mr * N + col] = v;
            }
        }
    }
}

// ---------------------------------------------------------------------------
// cam_mask[k,n] = any(bev_mask[k,0,n,:] > 0); count[n] = max(1, sum_k mask)
// ---------------------------------------------------------------------------
__global__ void masks_kernel(const int* __restrict__ bev, float* __restrict__ cam_mask,
                             float* __restrict__ cnt) {
    int n = blockIdx.x * blockDim.x + threadIdx.x;
    if (n >= NQ) return;
    float c = 0.0f;
    for (int k = 0; k < CAMS; ++k) {
        const int* p = bev + ((size_t)k * NQ + n) * 4;
        float m = (p[0] + p[1] + p[2] + p[3]) > 0 ? 1.0f : 0.0f;
        cam_mask[(size_t)k * NQ + n] = m;
        c += m;
    }
    cnt[n] = fmaxf(c, 1.0f);
}

// ---------------------------------------------------------------------------
// In-place softmax over groups of 16 (per head). ngroups = NQ*8.
// ---------------------------------------------------------------------------
__global__ void softmax16_kernel(float* __restrict__ aw, int ngroups) {
    int i = blockIdx.x * blockDim.x + threadIdx.x;
    if (i >= ngroups) return;
    float* p = aw + (size_t)i * 16;
    float mx = p[0];
#pragma unroll
    for (int j = 1; j < 16; ++j) mx = fmaxf(mx, p[j]);
    float s = 0.0f;
    float e[16];
#pragma unroll
    for (int j = 0; j < 16; ++j) { e[j] = __expf(p[j] - mx); s += e[j]; }
    float inv = 1.0f / s;
#pragma unroll
    for (int j = 0; j < 16; ++j) p[j] = e[j] * inv;
}

// ---------------------------------------------------------------------------
// Deformable bilinear sampling. block = (cam, query); thread = (head, dim).
// vproj is L2-resident (47MB < 192MB L2) -> gathers hit cache.
// ---------------------------------------------------------------------------
__global__ void msda_sample_kernel(const float* __restrict__ vproj,
                                   const float* __restrict__ off,
                                   const float* __restrict__ aw,
                                   const float* __restrict__ ref,
                                   float* __restrict__ out) {
    const int HH[4] = {60, 30, 15, 8};
    const int WW[4] = {100, 50, 25, 13};
    const int ST[4] = {0, 6000, 7500, 7875};

    int bq  = blockIdx.x;              // cam*NQ + q
    int cam = bq / NQ;
    int q   = bq - cam * NQ;
    int head = threadIdx.x >> 5;
    int d    = threadIdx.x & 31;
    int ch   = head * 32 + d;

    const float* offp = off + (size_t)q * DM + head * 32;   // (lvl, pt, 2)
    const float* awp  = aw  + (size_t)q * 128 + head * 16;  // (lvl, pt)
    const float* refp = ref + ((size_t)cam * NQ + q) * 8;   // (lvl, 2)

    float acc = 0.0f;
#pragma unroll
    for (int lvl = 0; lvl < 4; ++lvl) {
        int H = HH[lvl], W = WW[lvl];
        const float* vl = vproj + ((size_t)cam * LTOT + ST[lvl]) * DM;
        float rx = refp[lvl * 2 + 0];
        float ry = refp[lvl * 2 + 1];
#pragma unroll
        for (int pt = 0; pt < 4; ++pt) {
            float ox = offp[(lvl * 4 + pt) * 2 + 0];
            float oy = offp[(lvl * 4 + pt) * 2 + 1];
            float x = (rx + ox / (float)W) * (float)W - 0.5f;
            float y = (ry + oy / (float)H) * (float)H - 0.5f;
            float x0f = floorf(x), y0f = floorf(y);
            int x0 = (int)x0f, y0 = (int)y0f;
            float lx = x - x0f, ly = y - y0f;
            int x1 = x0 + 1, y1 = y0 + 1;
            float v00 = 0.f, v01 = 0.f, v10 = 0.f, v11 = 0.f;
            if (x0 >= 0 && x0 < W && y0 >= 0 && y0 < H) v00 = vl[((size_t)y0 * W + x0) * DM + ch];
            if (x1 >= 0 && x1 < W && y0 >= 0 && y0 < H) v01 = vl[((size_t)y0 * W + x1) * DM + ch];
            if (x0 >= 0 && x0 < W && y1 >= 0 && y1 < H) v10 = vl[((size_t)y1 * W + x0) * DM + ch];
            if (x1 >= 0 && x1 < W && y1 >= 0 && y1 < H) v11 = vl[((size_t)y1 * W + x1) * DM + ch];
            float samp = v00 * (1.f - lx) * (1.f - ly) + v01 * lx * (1.f - ly)
                       + v10 * (1.f - lx) * ly         + v11 * lx * ly;
            acc += awp[lvl * 4 + pt] * samp;
        }
    }
    out[(size_t)bq * DM + ch] = acc;
}

// ---------------------------------------------------------------------------
// slots = sum_k mask*(q + proj); x = LN(slots/count + q). One block per query.
// ---------------------------------------------------------------------------
__global__ void reduce_ln1_kernel(const float* __restrict__ q, const float* __restrict__ proj,
                                  const float* __restrict__ cam_mask, const float* __restrict__ cnt,
                                  const float* __restrict__ g, const float* __restrict__ b,
                                  float* __restrict__ x) {
    int n = blockIdx.x, c = threadIdx.x;
    float qv = q[(size_t)n * DM + c];
    float s = 0.0f;
    for (int k = 0; k < CAMS; ++k)
        s += cam_mask[(size_t)k * NQ + n] * (qv + proj[((size_t)k * NQ + n) * DM + c]);
    float val = s / cnt[n] + qv;

    __shared__ float red[DM];
    red[c] = val; __syncthreads();
    for (int t = DM / 2; t > 0; t >>= 1) { if (c < t) red[c] += red[c + t]; __syncthreads(); }
    float mu = red[0] / (float)DM; __syncthreads();
    float dv = val - mu;
    red[c] = dv * dv; __syncthreads();
    for (int t = DM / 2; t > 0; t >>= 1) { if (c < t) red[c] += red[c + t]; __syncthreads(); }
    float var = red[0] / (float)DM;
    x[(size_t)n * DM + c] = dv * rsqrtf(var + 1e-5f) * g[c] + b[c];
}

// ---------------------------------------------------------------------------
// out = LN(x + ffn). One block per query.
// ---------------------------------------------------------------------------
__global__ void ln2_kernel(const float* __restrict__ x, const float* __restrict__ ffn,
                           const float* __restrict__ g, const float* __restrict__ b,
                           float* __restrict__ out) {
    int n = blockIdx.x, c = threadIdx.x;
    float val = x[(size_t)n * DM + c] + ffn[(size_t)n * DM + c];
    __shared__ float red[DM];
    red[c] = val; __syncthreads();
    for (int t = DM / 2; t > 0; t >>= 1) { if (c < t) red[c] += red[c + t]; __syncthreads(); }
    float mu = red[0] / (float)DM; __syncthreads();
    float dv = val - mu;
    red[c] = dv * dv; __syncthreads();
    for (int t = DM / 2; t > 0; t >>= 1) { if (c < t) red[c] += red[c + t]; __syncthreads(); }
    float var = red[0] / (float)DM;
    out[(size_t)n * DM + c] = dv * rsqrtf(var + 1e-5f) * g[c] + b[c];
}

// ---------------------------------------------------------------------------
extern "C" void kernel_launch(void* const* d_in, const int* in_sizes, int n_in,
                              void* d_out, int out_size, void* d_ws, size_t ws_size,
                              hipStream_t stream) {
    const float* query   = (const float*)d_in[0];
    // d_in[1] (key) is unused by the reference computation.
    const float* value   = (const float*)d_in[2];
    const float* refpts  = (const float*)d_in[3];
    const int*   bev     = (const int*)d_in[6];
    const float* W_value = (const float*)d_in[7];
    const float* b_value = (const float*)d_in[8];
    const float* W_off   = (const float*)d_in[9];
    const float* b_off   = (const float*)d_in[10];
    const float* W_attn  = (const float*)d_in[11];
    const float* b_attn  = (const float*)d_in[12];
    const float* W_out   = (const float*)d_in[13];
    const float* b_out   = (const float*)d_in[14];
    const float* ln1_g   = (const float*)d_in[15];
    const float* ln1_b   = (const float*)d_in[16];
    const float* W1      = (const float*)d_in[17];
    const float* b1      = (const float*)d_in[18];
    const float* W2      = (const float*)d_in[19];
    const float* b2      = (const float*)d_in[20];
    const float* ln2_g   = (const float*)d_in[21];
    const float* ln2_b   = (const float*)d_in[22];
    float* out = (float*)d_out;

    // ---- workspace carve ----
    char* w = (char*)d_ws;
    auto carve = [&](size_t bytes) -> char* {
        char* p = w; w += (bytes + 255) & ~(size_t)255; return p;
    };
    _Float16* fWv = (_Float16*)carve((size_t)DM * DM   * 2);
    _Float16* fWo = (_Float16*)carve((size_t)DM * DM   * 2);
    _Float16* fWa = (_Float16*)carve((size_t)DM * 128  * 2);
    _Float16* fWp = (_Float16*)carve((size_t)DM * DM   * 2);
    _Float16* fW1 = (_Float16*)carve((size_t)DM * DFFN * 2);
    _Float16* fW2 = (_Float16*)carve((size_t)DFFN * DM * 2);
    float* cam_mask = (float*)carve((size_t)CAMS * NQ * 4);
    float* cnt      = (float*)carve((size_t)NQ * 4);
    float* vproj    = (float*)carve((size_t)CAMS * LTOT * DM * 4); // reused: ffn_out
    float* offb     = (float*)carve((size_t)NQ * DM * 4);
    float* awb      = (float*)carve((size_t)NQ * 128 * 4);
    float* sampled  = (float*)carve((size_t)CAMS * NQ * DM * 4);
    float* proj     = (float*)carve((size_t)CAMS * NQ * DM * 4);  // reused: ffn hidden
    float* xbuf     = (float*)carve((size_t)NQ * DM * 4);
    float* hidden   = proj;     // dead after reduce_ln1
    float* ffnout   = vproj;    // dead after sampling

    auto gemm16 = [&](const float* A, const _Float16* Bf, const float* bias, float* C,
                      int M, int N, int K, int relu) {
        long strips = (long)((M + 15) / 16) * (N / 256);
        int blocks = (int)((strips + 7) / 8);
        wmma_gemm_kernel<16><<<blocks, 256, 0, stream>>>(A, Bf, bias, C, M, N, K, relu);
    };
    auto gemm8 = [&](const float* A, const _Float16* Bf, const float* bias, float* C,
                     int M, int N, int K, int relu) {
        long strips = (long)((M + 15) / 16) * (N / 128);
        int blocks = (int)((strips + 7) / 8);
        wmma_gemm_kernel<8><<<blocks, 256, 0, stream>>>(A, Bf, bias, C, M, N, K, relu);
    };

    // 1. swizzle weights to f16 B-fragments
    swizzle_w_kernel<<<(DM * DM)   / 256, 256, 0, stream>>>(W_value, fWv, DM, DM);
    swizzle_w_kernel<<<(DM * DM)   / 256, 256, 0, stream>>>(W_off,   fWo, DM, DM);
    swizzle_w_kernel<<<(DM * 128)  / 256, 256, 0, stream>>>(W_attn,  fWa, DM, 128);
    swizzle_w_kernel<<<(DM * DM)   / 256, 256, 0, stream>>>(W_out,   fWp, DM, DM);
    swizzle_w_kernel<<<(DM * DFFN) / 256, 256, 0, stream>>>(W1,      fW1, DM, DFFN);
    swizzle_w_kernel<<<(DFFN * DM) / 256, 256, 0, stream>>>(W2,      fW2, DFFN, DM);

    // 2. visibility masks / counts
    masks_kernel<<<(NQ + 255) / 256, 256, 0, stream>>>(bev, cam_mask, cnt);

    // 3. value projection (all cams): full-N strip => each value element
    //    loaded exactly once (dominant 49MB HBM stream).
    gemm16(value, fWv, b_value, vproj, CAMS * LTOT, DM, DM, 0);

    // 4. sampling offsets + attention logits (once; identical across cams)
    gemm16(query, fWo, b_off,  offb, NQ, DM,  DM, 0);
    gemm8 (query, fWa, b_attn, awb,  NQ, 128, DM, 0);

    // 5. softmax over 16 points per head (in place)
    softmax16_kernel<<<(NQ * 8 + 255) / 256, 256, 0, stream>>>(awb, NQ * 8);

    // 6. deformable bilinear sampling -> (CAMS*NQ x 256)
    msda_sample_kernel<<<CAMS * NQ, 256, 0, stream>>>(vproj, offb, awb, refpts, sampled);

    // 7. output projection (+b_out)
    gemm16(sampled, fWp, b_out, proj, CAMS * NQ, DM, DM, 0);

    // 8. masked scatter-reduce + residual + LayerNorm1
    reduce_ln1_kernel<<<NQ, DM, 0, stream>>>(query, proj, cam_mask, cnt, ln1_g, ln1_b, xbuf);

    // 9. FFN: relu(x @ W1 + b1) @ W2 + b2
    gemm16(xbuf,   fW1, b1, hidden, NQ, DFFN, DM,   1);
    gemm16(hidden, fW2, b2, ffnout, NQ, DM,   DFFN, 0);

    // 10. residual + LayerNorm2 -> output
    ln2_kernel<<<NQ, DM, 0, stream>>>(xbuf, ffnout, ln2_g, ln2_b, out);
}
